// TorchCE_74354473828837
// MI455X (gfx1250) — matrix operand order repaired and checked by reading, not compiled, not run on hardware
//
#include <hip/hip_runtime.h>
#include <hip/hip_bf16.h>
#include <math.h>

// Fused linear + softcap + label-smoothed CE + z-loss for
//   X[4096,2048] f32 @ W[2048,32000] f32, streamed with online sum-exp
// (softcap bounds logits to [-30,30] so no max tracking is needed).
// GEMM runs as split-bf16 (hi/lo) WMMA: acc += Ah*Bh + Ah*Bl + Al*Bh.
// W and X are pre-converted once into bf16 hi/lo planes in WMMA fragment
// order; the main kernel stages its A tile via the Tensor Data Mover
// (tensor_load_to_lds + s_wait_tensorcnt) and runs a pure load+WMMA loop.

#define CE_N 4096
#define CE_D 2048
#define CE_V 32000
#define CE_SOFTCAP 30.0f
#define CE_SMOOTH 0.1f
#define CE_ZSCALE 1e-4f
#define CE_IGNORE (-100)

#define ROWS_PER_WG 32       // two 16-row M tiles
#define WAVES 8              // 256 threads
#define KTILES (CE_D / 32)   // 64
#define COLTILES (CE_V / 16) // 2000
#define COLPAIRS (COLTILES / 2) // 1000 (each wave takes a 32-col pair)
#define VSPLIT 2
#define ROWGROUPS (CE_N / ROWS_PER_WG) // 128

#define FRAG_USH 16                       // ushorts per lane per fragment
#define TILE_USH ((size_t)KTILES * 32 * FRAG_USH) // ushorts per col-tile (32768)
#define WPLANE ((size_t)COLTILES * TILE_USH)      // ushorts per W bf16 plane
#define ARG_USH ((size_t)2 * KTILES * 32 * FRAG_USH) // ushorts per rowgroup A chunk (65536)
#define APLANE ((size_t)ROWGROUPS * ARG_USH)         // ushorts per A bf16 plane
#define A_TDM_ELEMS 16384   // ARG_USH*2 bytes / 8-byte elements

typedef __attribute__((ext_vector_type(16))) __bf16 v16bf;
typedef __attribute__((ext_vector_type(8))) float v8f;
typedef __attribute__((ext_vector_type(4))) unsigned int v4u;
typedef __attribute__((ext_vector_type(8))) int v8i;
typedef __attribute__((ext_vector_type(4))) int v4i;

__device__ __forceinline__ void f32_to_bf16x2(float v, unsigned short& h, unsigned short& l) {
  unsigned int u = __float_as_uint(v);
  unsigned int hr = (u + 0x7FFFu + ((u >> 16) & 1u)) >> 16; // RNE
  h = (unsigned short)hr;
  float hf = __uint_as_float(hr << 16);
  unsigned int u2 = __float_as_uint(v - hf);
  l = (unsigned short)((u2 + 0x7FFFu + ((u2 >> 16) & 1u)) >> 16);
}

__device__ __forceinline__ v8f wmma_bf16(v16bf a, v16bf b, v8f c) {
  return __builtin_amdgcn_wmma_f32_16x16x32_bf16(false, a, false, b, (short)0, c, false, false);
}

// 1-D TDM copy: `elems` 8-byte elements from global -> LDS (D# per ISA 8.3/8.4).
// This toolchain exposes the 6-arg tensor_load_to_lds builtin
// (g0:u32x4, g1:i32x8, g2:i32x4, g3:i32x4, g4:i32x8, cpol:i32).
__device__ __forceinline__ void tdm_load_1d(unsigned lds_addr, const void* gptr,
                                            unsigned elems) {
  unsigned long long ga = (unsigned long long)(uintptr_t)gptr;
  v4u g0;
  g0[0] = 1u;                                   // count=1, user descriptor
  g0[1] = lds_addr;                             // LDS byte address
  g0[2] = (unsigned)(ga & 0xFFFFFFFFu);         // global_addr[31:0]
  g0[3] = (unsigned)((ga >> 32) & 0x01FFFFFFu)  // global_addr[56:32]
          | (2u << 30);                         // type=2 ("image")
  v8i g1;
  g1[0] = (int)(3u << 16);                      // data_size=3 (8 bytes)
  g1[1] = (int)((elems & 0xFFFFu) << 16);       // tensor_dim0[15:0]
  g1[2] = (int)(((elems >> 16) & 0xFFFFu)       // tensor_dim0[31:16]
          | (1u << 16));                        // tensor_dim1 = 1
  g1[3] = (int)((elems & 0xFFFFu) << 16);       // tile_dim0 = elems
  g1[4] = 1;                                    // tile_dim1 = 1, tile_dim2 = 0
  g1[5] = (int)elems;                           // tensor_dim0_stride[31:0]
  g1[6] = 0;
  g1[7] = 0;
  v4i gz4 = {0, 0, 0, 0};
  v8i gz8 = {0, 0, 0, 0, 0, 0, 0, 0};
  __builtin_amdgcn_tensor_load_to_lds(g0, g1, gz4, gz4, gz8, 0);
}

// ---------------------------------------------------------------------------
// Pre-pass 1: convert W (f32, [D][V] row-major) into bf16 hi/lo planes stored
// in B-fragment order: plane[((t*KTILES + kt)*32 + lane)*16 + j] where
//   lane<16  -> column n=t*16+lane,    K = kt*32 + j      (j=0..15)
//   lane>=16 -> column n=t*16+lane-16, K = kt*32 + 16 + j
// ---------------------------------------------------------------------------
__global__ __launch_bounds__(256) void ce_convert_w(
    const float* __restrict__ w, unsigned short* __restrict__ whi,
    unsigned short* __restrict__ wlo) {
  int gwave = (int)((blockIdx.x * 256 + threadIdx.x) >> 5);
  int lane = threadIdx.x & 31;
  if (gwave >= COLTILES * KTILES) return;
  int t = gwave / KTILES, kt = gwave - t * KTILES;
  int n = t * 16 + (lane & 15);
  int kbase = kt * 32 + (lane >> 4) * 16;
  const float* p = w + (size_t)kbase * CE_V + n;
  unsigned short h[16], l[16];
#pragma unroll
  for (int j = 0; j < 16; ++j) f32_to_bf16x2(p[(size_t)j * CE_V], h[j], l[j]);
  size_t off = ((size_t)gwave * 32 + lane) * 16;
  __builtin_memcpy(whi + off, h, 32);
  __builtin_memcpy(wlo + off, l, 32);
}

// ---------------------------------------------------------------------------
// Pre-pass 2: convert X (f32, [N][D]) into bf16 hi/lo planes in A-fragment
// order, one contiguous 128KB chunk per rowgroup: [rg][mt][kt][lane][j].
// A order used here (contiguous k): lane<16 -> m=lane, K={0..15};
//                        lane>=16 -> m=lane-16, K={16..31}.
// ---------------------------------------------------------------------------
__global__ __launch_bounds__(256) void ce_convert_a(
    const float* __restrict__ x, unsigned short* __restrict__ ahi,
    unsigned short* __restrict__ alo) {
  int gwave = (int)((blockIdx.x * 256 + threadIdx.x) >> 5); // (rg*2+mt)*KTILES+kt
  int lane = threadIdx.x & 31;
  if (gwave >= ROWGROUPS * 2 * KTILES) return;
  int kt = gwave % KTILES;
  int mtg = gwave / KTILES;
  int rg = mtg >> 1, mt = mtg & 1;
  int m = rg * ROWS_PER_WG + mt * 16 + (lane & 15);
  int kadd = (lane >> 4) * 16;
  const float* p = x + (size_t)m * CE_D + kt * 32 + kadd;
  unsigned short h[16], l[16];
#pragma unroll
  for (int j = 0; j < 16; ++j) {
    int ko = j;
    f32_to_bf16x2(p[ko], h[j], l[j]);
  }
  size_t off = ((size_t)gwave * 32 + lane) * 16;
  __builtin_memcpy(ahi + off, h, 32);
  __builtin_memcpy(alo + off, l, 32);
}

// ---------------------------------------------------------------------------
// Main fused GEMM + streaming CE-stats kernel.
// ---------------------------------------------------------------------------
__global__ __launch_bounds__(256) void ce_main_kernel(
    const unsigned short* __restrict__ ahi, const unsigned short* __restrict__ alo,
    const unsigned short* __restrict__ whi, const unsigned short* __restrict__ wlo,
    const int* __restrict__ targets, float* __restrict__ part) {
  // A tile in WMMA A-fragment order: [mtile][ktile][lane][16 bf16] = 128KB/plane
  __shared__ __align__(16) unsigned short Ahi[2][KTILES][32][16];
  __shared__ __align__(16) unsigned short Alo[2][KTILES][32][16];
  __shared__ float wEsum[WAVES][ROWS_PER_WG];
  __shared__ float wZsum[WAVES][ROWS_PER_WG];
  __shared__ int   tIdx[ROWS_PER_WG];
  __shared__ float tLog[ROWS_PER_WG];

  const int tid  = threadIdx.x;
  const int wave = tid >> 5;
  const int lane = tid & 31;
  const int rg   = blockIdx.x;          // row group
  const int sp   = blockIdx.y;          // V split
  const int rowbase = rg * ROWS_PER_WG;

  if (tid < ROWS_PER_WG) {
    tIdx[tid] = targets[rowbase + tid];
    tLog[tid] = 0.0f;
  }
  // Opaque (never-taken) store: the TDM write below is invisible to alias
  // analysis, so keep the compiler from treating Ahi/Alo as never-written.
  if (tid == 0xFFFF) { Ahi[0][0][0][0] = 1; Alo[0][0][0][0] = 1; }

  // ---- Stage A tile via Tensor Data Mover: 2 x 128KB global -> LDS ----
  if (wave == 0) {
    unsigned lds_hi = (unsigned)(uintptr_t)(void*)&Ahi[0][0][0][0];
    unsigned lds_lo = (unsigned)(uintptr_t)(void*)&Alo[0][0][0][0];
    tdm_load_1d(lds_hi, ahi + (size_t)rg * ARG_USH, A_TDM_ELEMS);
    tdm_load_1d(lds_lo, alo + (size_t)rg * ARG_USH, A_TDM_ELEMS);
    __builtin_amdgcn_s_wait_tensorcnt(0);
  }
  __syncthreads();

  const int nsub  = lane & 15;   // column within 16-wide tile
  const int khalf = lane >> 4;   // C/D rows: lanes 16-31 carry m+8

  // Per-lane target columns for the 16 (mt,r) rows this lane sees.
  int tcol[16];
#pragma unroll
  for (int i = 0; i < 16; ++i) {
    int rowl = (i >> 3) * 16 + (i & 7) + khalf * 8;
    tcol[i] = tIdx[rowl];
  }

  // Per-lane running stats (softcap bounds z, so plain sum-exp is safe).
  float es[16], zsm[16];
#pragma unroll
  for (int i = 0; i < 16; ++i) { es[i] = 0.0f; zsm[i] = 0.0f; }

  // Each wave takes a pair of adjacent 16-col tiles (32 columns) per pass.
  for (int q = sp * WAVES + wave; q < COLPAIRS; q += WAVES * VSPLIT) {
    const int n0 = q * 32;
    v8f acc00 = {}, acc10 = {};  // tile0: mtile0 / mtile1
    v8f acc01 = {}, acc11 = {};  // tile1: mtile0 / mtile1
    const unsigned short* pbh = whi + (size_t)(2 * q) * TILE_USH + (size_t)lane * FRAG_USH;
    const unsigned short* pbl = wlo + (size_t)(2 * q) * TILE_USH + (size_t)lane * FRAG_USH;

#pragma unroll 2
    for (int kt = 0; kt < KTILES; ++kt) {
      v16bf Bh0, Bl0, Bh1, Bl1, Ah0, Al0, Ah1, Al1;
      // tile1 fragments sit at a constant +TILE_USH (64KB) immediate offset
      __builtin_memcpy(&Bh0, pbh + (size_t)kt * 512, 32);
      __builtin_memcpy(&Bh1, pbh + (size_t)kt * 512 + TILE_USH, 32);
      __builtin_memcpy(&Bl0, pbl + (size_t)kt * 512, 32);
      __builtin_memcpy(&Bl1, pbl + (size_t)kt * 512 + TILE_USH, 32);
      __builtin_memcpy(&Ah0, &Ahi[0][kt][lane][0], 32);
      __builtin_memcpy(&Al0, &Alo[0][kt][lane][0], 32);
      __builtin_memcpy(&Ah1, &Ahi[1][kt][lane][0], 32);
      __builtin_memcpy(&Al1, &Alo[1][kt][lane][0], 32);

      acc00 = wmma_bf16(Ah0, Bh0, acc00);
      acc00 = wmma_bf16(Ah0, Bl0, acc00);
      acc00 = wmma_bf16(Al0, Bh0, acc00);
      acc10 = wmma_bf16(Ah1, Bh0, acc10);
      acc10 = wmma_bf16(Ah1, Bl0, acc10);
      acc10 = wmma_bf16(Al1, Bh0, acc10);
      acc01 = wmma_bf16(Ah0, Bh1, acc01);
      acc01 = wmma_bf16(Ah0, Bl1, acc01);
      acc01 = wmma_bf16(Al0, Bh1, acc01);
      acc11 = wmma_bf16(Ah1, Bh1, acc11);
      acc11 = wmma_bf16(Ah1, Bl1, acc11);
      acc11 = wmma_bf16(Al1, Bh1, acc11);
    }

    // ---- streaming epilogue: softcap + per-lane accumulation ----
    // C/D layout: VGPR r, lanes 0-15 -> (m=r, n), lanes 16-31 -> (m=r+8, n).
#pragma unroll
    for (int tt = 0; tt < 2; ++tt) {
      const int ncol = n0 + tt * 16 + nsub;
#pragma unroll
      for (int mt = 0; mt < 2; ++mt) {
        v8f acc = tt ? (mt ? acc11 : acc01) : (mt ? acc10 : acc00);
#pragma unroll
        for (int r = 0; r < 8; ++r) {
          float a = acc[r] * (1.0f / CE_SOFTCAP);
          // branchless tanh: 1 - 2/(1+e^{2a}) (exact in both saturation limits)
          float e2 = __expf(2.0f * a);
          float z = CE_SOFTCAP * (1.0f - 2.0f / (e2 + 1.0f));
          int i = mt * 8 + r;
          es[i] += __expf(z);
          zsm[i] += z;
          if (ncol == tcol[i]) tLog[mt * 16 + r + khalf * 8] = z;
        }
      }
    }
  }

  // ---- deterministic end-of-kernel reduction ----
#pragma unroll
  for (int i = 0; i < 16; ++i) {
    float e = es[i], z = zsm[i];
#pragma unroll
    for (int msk = 1; msk <= 8; msk <<= 1) {
      e += __shfl_xor(e, msk, 32);
      z += __shfl_xor(z, msk, 32);
    }
    if (nsub == 0) {
      int rowl = (i >> 3) * 16 + (i & 7) + khalf * 8;
      wEsum[wave][rowl] = e;
      wZsum[wave][rowl] = z;
    }
  }
  __syncthreads();

  if (tid < ROWS_PER_WG) {
    float E = 0.0f, Z = 0.0f;
#pragma unroll
    for (int w = 0; w < WAVES; ++w) { E += wEsum[w][tid]; Z += wZsum[w][tid]; }
    float* o = part + (((size_t)rg * VSPLIT + sp) * ROWS_PER_WG + tid) * 4;
    o[0] = E; o[1] = Z; o[2] = tLog[tid]; o[3] = 0.0f;
  }
}

__global__ void ce_combine_kernel(const float* __restrict__ part,
                                  const int* __restrict__ targets,
                                  float* __restrict__ ws2) {
  int rg = blockIdx.x;
  int row = threadIdx.x;  // 32 threads = one wave
  float E = 0.0f, Z = 0.0f, TL = 0.0f;
  for (int sp = 0; sp < VSPLIT; ++sp) {
    const float* o = part + (((size_t)rg * VSPLIT + sp) * ROWS_PER_WG + row) * 4;
    E += o[0];
    Z += o[1];
    TL += o[2];  // only the split containing the target column contributes
  }
  float lse = __logf(E);
  int tv = targets[rg * ROWS_PER_WG + row];
  bool valid = (tv != CE_IGNORE);
  float nll = lse - TL;
  float smooth = lse - Z * (1.0f / CE_V);
  float ce = (1.0f - CE_SMOOTH) * nll + CE_SMOOTH * smooth;
  float zl = CE_ZSCALE * lse * lse;
  float c = valid ? (ce + zl) : 0.0f;
  float n = valid ? 1.0f : 0.0f;
#pragma unroll
  for (int msk = 1; msk <= 16; msk <<= 1) {
    c += __shfl_xor(c, msk, 32);
    n += __shfl_xor(n, msk, 32);
  }
  if (row == 0) {
    ws2[rg] = c;
    ws2[ROWGROUPS + rg] = n;
  }
}

__global__ void ce_finalize_kernel(const float* __restrict__ ws2, float* __restrict__ out) {
  if (threadIdx.x == 0 && blockIdx.x == 0) {
    float c = 0.0f, n = 0.0f;
    for (int i = 0; i < ROWGROUPS; ++i) { c += ws2[i]; n += ws2[ROWGROUPS + i]; }
    out[0] = c / n;
  }
}

extern "C" void kernel_launch(void* const* d_in, const int* in_sizes, int n_in,
                              void* d_out, int out_size, void* d_ws, size_t ws_size,
                              hipStream_t stream) {
  const float* input   = (const float*)d_in[0];
  const float* weight  = (const float*)d_in[1];
  const int*   targets = (const int*)d_in[2];
  // d_in[3] is bias == 0 (falsy) -> ignored, matching the reference branch.
  (void)in_sizes; (void)n_in; (void)out_size; (void)ws_size;

  unsigned short* whi = (unsigned short*)d_ws;   // 131 MB W bf16-hi plane
  unsigned short* wlo = whi + WPLANE;            // 131 MB W bf16-lo plane
  unsigned short* ahi = wlo + WPLANE;            // 16 MB A bf16-hi plane
  unsigned short* alo = ahi + APLANE;            // 16 MB A bf16-lo plane
  float* part = (float*)(alo + APLANE);          // 128*2*32*4 floats
  float* ws2  = part + (size_t)ROWGROUPS * VSPLIT * ROWS_PER_WG * 4;

  int convw_blocks = (COLTILES * KTILES + WAVES - 1) / WAVES;
  ce_convert_w<<<convw_blocks, 256, 0, stream>>>(weight, whi, wlo);
  int conva_blocks = (ROWGROUPS * 2 * KTILES + WAVES - 1) / WAVES;
  ce_convert_a<<<conva_blocks, 256, 0, stream>>>(input, ahi, alo);

  dim3 grid(ROWGROUPS, VSPLIT);
  ce_main_kernel<<<grid, 256, 0, stream>>>(ahi, alo, whi, wlo, targets, part);
  ce_combine_kernel<<<ROWGROUPS, 32, 0, stream>>>(part, targets, ws2);
  ce_finalize_kernel<<<1, 32, 0, stream>>>(ws2, (float*)d_out);
}
